// ModularSAMHA_32220844654881
// MI455X (gfx1250) — matrix operand order repaired and checked by reading, not compiled, not run on hardware
//
#include <hip/hip_runtime.h>
#include <hip/hip_bf16.h>
#include <stdint.h>

#define NHEADS 8
#define DK 32
#define DMODEL 256
#define CCH 256
#define HDIM 48
#define WDIM 48
#define HWN 2304
#define BB 2

typedef __bf16 bf16_t;
typedef __attribute__((ext_vector_type(16))) __bf16 v16bf;
typedef __attribute__((ext_vector_type(8)))  float  v8f;
typedef __attribute__((ext_vector_type(4)))  unsigned int v4u;
typedef __attribute__((ext_vector_type(8)))  int v8i;
typedef __attribute__((ext_vector_type(4)))  int v4i;

// A-fragment (16x32 bf16): lane holds one M row; K = 16*hh + j (contiguous 16)
//   (hh = lane>>4)
static __device__ __forceinline__ v16bf load_fragA(const bf16_t* row, int hh) {
  union { uint4 u[2]; v16bf v; } t;
  t.u[0] = *(const uint4*)(row + 16 * hh);
  t.u[1] = *(const uint4*)(row + 16 * hh + 8);
  return t.v;
}
// B-fragment (32x16 bf16): lane holds one N column; K = 16*hh + j (contiguous 16)
static __device__ __forceinline__ v16bf load_fragB(const bf16_t* row, int hh) {
  union { uint4 u[2]; v16bf v; } t;
  const bf16_t* p = row + 16 * hh;
  t.u[0] = *(const uint4*)(p);
  t.u[1] = *(const uint4*)(p + 8);
  return t.v;
}
static __device__ __forceinline__ v8f wmma_bf16(v16bf a, v16bf b, v8f c) {
  return __builtin_amdgcn_wmma_f32_16x16x32_bf16(false, a, false, b, (short)0, c,
                                                 false, false);
}

// --------------------------------------------------------------- TDM tile load
// 2D tile: 16 dwords (64B) x 32 rows, global row pitch `pitch_dwords`,
// LDS destination padded +4 dwords per 16 dwords (row pitch 80B -> 16-lane
// ds_load_b128 fragment reads are bank-conflict free: stride 20 banks).
// Descriptor layout per CDNA5 ISA.
static __device__ __forceinline__ void tdm_tile_load(uint32_t lds_byte,
                                                     const void* gptr,
                                                     uint32_t pitch_dwords) {
  uint64_t ga = (uint64_t)(uintptr_t)gptr;
  v4u g0 = { 1u,                                  // count=1 (valid user D#)
             lds_byte,                            // lds_addr (bytes)
             (uint32_t)ga,                        // global_addr[31:0]
             (uint32_t)((ga >> 32) & 0x1FFFFFFu) | (2u << 30) }; // addr[56:32]|type=2
  v8i g1;
  g1[0] = (int)((2u << 16)      // data_size = 4 bytes
                | (1u << 20)    // pad_enable
                | (3u << 22)    // pad_interval: every 16 dwords
                | (3u << 25));  // pad_amount: 4 dwords
  g1[1] = (int)(16u << 16);     // tensor_dim0 = 16 dwords  (bits 79:48 low half)
  g1[2] = (int)(32u << 16);     // tensor_dim0 hi=0 ; tensor_dim1 = 32 rows
  g1[3] = (int)(16u << 16);     // tensor_dim1 hi=0 ; tile_dim0 = 16 dwords
  g1[4] = 32;                   // tile_dim1 = 32 ; tile_dim2 = 0
  g1[5] = (int)pitch_dwords;    // tensor_dim0_stride[31:0]
  g1[6] = 0;                    // dim0_stride hi ; dim1_stride lo
  g1[7] = 0;
  v4i z4 = {0, 0, 0, 0};
#if __clang_major__ >= 23
  v8i z8 = {};
  __builtin_amdgcn_tensor_load_to_lds(g0, g1, z4, z4, z8, 0);
#else
  __builtin_amdgcn_tensor_load_to_lds(g0, g1, z4, z4, 0);
#endif
}

// ---------------------------------------------------------------- setup kernels
__global__ void pe_kernel(float* pe) {
  int i = blockIdx.x * 256 + threadIdx.x;
  if (i >= DMODEL * HWN) return;
  int d = i / HWN, nn = i % HWN;
  int h_ = nn / WDIM, w_ = nn % WDIM;
  int dd = (d < 128) ? d : d - 128;
  float coord = (d < 128) ? (float)w_ : (float)h_;
  int j = dd >> 1;
  float inv = __expf(-(float)(2 * j) * (9.210340371976184f / 128.0f));
  float ang = coord * inv;
  pe[i] = (dd & 1) ? __cosf(ang) : __sinf(ang);
}

__global__ void dm_kernel(float* dm) {
  int i = blockIdx.x * 256 + threadIdx.x;
  if (i >= HWN * HWN) return;
  int a = i / HWN, c = i % HWN;
  float dy = (float)(a / WDIM - c / WDIM);
  float dx = (float)(a % WDIM - c % WDIM);
  dm[i] = __expf(-__fsqrt_rn(dy * dy + dx * dx)); // SIGMA = 1.0
}

__global__ void cvt_bf16_kernel(const float* src, bf16_t* dst, int n) {
  int i = blockIdx.x * 256 + threadIdx.x;
  if (i < n) dst[i] = (bf16_t)src[i];
}

// x (B,C,HW) f32 -> xbf (B,HW,C) bf16
__global__ void transpose_x_kernel(const float* x, bf16_t* xbf) {
  int i = blockIdx.x * 256 + threadIdx.x;
  if (i >= BB * HWN * CCH) return;
  int b = i / (HWN * CCH);
  int r = i % (HWN * CCH);
  int nn = r / CCH, c = r % CCH;
  xbf[i] = (bf16_t)x[((size_t)b * CCH + c) * HWN + nn];
}

// ---------------------------------------------------------------- QKV projection
__global__ void __launch_bounds__(128)
proj_kernel(const bf16_t* __restrict__ xbf,
            const bf16_t* __restrict__ Wqb, const bf16_t* __restrict__ Wkb,
            const bf16_t* __restrict__ Wvb,
            const float* __restrict__ bq, const float* __restrict__ bk,
            const float* __restrict__ bv, const float* __restrict__ pe,
            bf16_t* __restrict__ Qbf, bf16_t* __restrict__ Kbf,
            bf16_t* __restrict__ Vbf) {
  const int lane = threadIdx.x & 31;
  const int laneLo = lane & 15;
  const int hh = lane >> 4;
  const int tile = blockIdx.x * 4 + (threadIdx.x >> 5); // 0..4607
  const int b = tile / (16 * 144);
  const int rem = tile % (16 * 144);
  const int o0 = (rem / 144) * 16;
  const int n0 = (rem % 144) * 16;

  const int n = n0 + laneLo;
  const int oA = o0 + laneLo;
  const bf16_t* xrow = xbf + ((size_t)b * HWN + n) * CCH;

  v8f aq = {}, ak = {}, av = {};
  for (int c0 = 0; c0 < CCH; c0 += 32) {
    v16bf bx = load_fragB(xrow + c0, hh);
    v16bf wq = load_fragA(Wqb + (size_t)oA * CCH + c0, hh);
    v16bf wk = load_fragA(Wkb + (size_t)oA * CCH + c0, hh);
    v16bf wv = load_fragA(Wvb + (size_t)oA * CCH + c0, hh);
    aq = wmma_bf16(wq, bx, aq);
    ak = wmma_bf16(wk, bx, ak);
    av = wmma_bf16(wv, bx, av);
  }
  const float qscale = 0.17677669529663687f; // 1/sqrt(32)
#pragma unroll
  for (int v = 0; v < 8; ++v) {
    int o = o0 + v + hh * 8;
    float pev = pe[(size_t)o * HWN + n];
    float qv = (aq[v] + bq[o] + pev) * qscale;
    float kv = ak[v] + bk[o] + pev;
    float vv = av[v] + bv[o];
    int head = o >> 5, dloc = o & 31;
    size_t qk = (((size_t)(b * NHEADS + head)) * HWN + n) * DK + dloc;
    Qbf[qk] = (bf16_t)qv;
    Kbf[qk] = (bf16_t)kv;
    Vbf[((size_t)b * DMODEL + o) * HWN + n] = (bf16_t)vv;
  }
}

// ---------------------------------------------------------------- flash attention
// One workgroup (4 waves) owns 64 query rows of one (b,h) and streams all 2304
// keys in 32-wide blocks. K/V tiles are DMA'd to LDS once per workgroup by the
// Tensor Data Mover (double-buffered, s_wait_tensorcnt + barriers), cutting
// K/V L2 traffic 4x vs per-wave loads. S^T = K*Q^T so each lane owns one
// softmax row; the P^T C-fragment feeds the O^T = V^T*P^T B-fragment after a
// single shfl_xor(16) half-exchange.
__global__ void __launch_bounds__(128)
attn_kernel(const bf16_t* __restrict__ Qbf, const bf16_t* __restrict__ Kbf,
            const bf16_t* __restrict__ Vbf, const float* __restrict__ dm,
            const float* __restrict__ lambda_p, bf16_t* __restrict__ Obf) {
  // 32 rows x 40 bf16 (80B padded pitch) per tile, double buffered
  __shared__ __attribute__((aligned(16))) bf16_t sK[2][32 * 40];
  __shared__ __attribute__((aligned(16))) bf16_t sV[2][32 * 40];

  const int lane = threadIdx.x & 31;
  const int laneLo = lane & 15;
  const int hh = lane >> 4;
  const int wave = threadIdx.x >> 5;
  const int bh = blockIdx.x / 36;   // 16 (b,h) pairs
  const int nblk = blockIdx.x % 36; // 36 blocks of 64 query rows
  const int n0 = nblk * 64 + wave * 16;
  const int b = bh >> 3;
  const int h = bh & 7;
  const float lambda = *lambda_p;

  const bf16_t* Qh = Qbf + (size_t)bh * HWN * DK;
  const bf16_t* Kh = Kbf + (size_t)bh * HWN * DK;
  const bf16_t* Vh = Vbf + ((size_t)b * DMODEL + h * DK) * HWN;

  const int n = n0 + laneLo;
  const v16bf bq = load_fragB(Qh + (size_t)n * DK, hh); // Q^T, K-dim = d_k = 32

  // prologue: DMA tile 0 into buffer 0
  if (wave == 0) {
    tdm_tile_load((uint32_t)(uintptr_t)&sK[0][0], Kh, DK / 2);
    tdm_tile_load((uint32_t)(uintptr_t)&sV[0][0], Vh, HWN / 2);
  }

  v8f o0 = {}, o1 = {};
  float mrow = -1e30f, lrow = 0.0f;
  int cur = 0;

  for (int m0 = 0; m0 < HWN; m0 += 32) {
    __syncthreads(); // everyone done reading buf[cur^1] (previous iteration)
    if (wave == 0) {
      int mn = m0 + 32;
      if (mn > HWN - 32) mn = HWN - 32; // clamp: redundant reload, never OOB
      tdm_tile_load((uint32_t)(uintptr_t)&sK[cur ^ 1][0],
                    Kh + (size_t)mn * DK, DK / 2);
      tdm_tile_load((uint32_t)(uintptr_t)&sV[cur ^ 1][0], Vh + mn, HWN / 2);
      __builtin_amdgcn_s_wait_tensorcnt(2); // tile for buf[cur] has landed
    }
    __syncthreads(); // publish buf[cur] to all waves

    // prefetch next dm tile rows (largest streaming consumer, L2-resident)
    {
      int pm = m0 + 32 + lane;
      if (pm > HWN - 1) pm = HWN - 1;
      __builtin_prefetch(&dm[(size_t)pm * HWN + n], 0, 1);
    }

    // S^T tiles from LDS (ds_load_b128, conflict-free 80B pitch)
    v16bf ak0 = load_fragA(&sK[cur][laneLo * 40], hh);
    v16bf ak1 = load_fragA(&sK[cur][(16 + laneLo) * 40], hh);
    v8f z = {};
    v8f s0 = wmma_bf16(ak0, bq, z);
    v8f s1 = wmma_bf16(ak1, bq, z);

    // distance-prior bias; element v of s0/s1 is (m = m0[+16] + v + 8*hh, n)
    float sv[16];
#pragma unroll
    for (int v = 0; v < 8; ++v) {
      int mg0 = m0 + v + hh * 8;
      int mg1 = mg0 + 16;
      sv[v]     = s0[v] + lambda * dm[(size_t)mg0 * HWN + n]; // dm symmetric
      sv[v + 8] = s1[v] + lambda * dm[(size_t)mg1 * HWN + n];
    }
    // streaming softmax over this lane's 16 m's + partner lane's 16
    float tmax = sv[0];
#pragma unroll
    for (int j = 1; j < 16; ++j) tmax = fmaxf(tmax, sv[j]);
    tmax = fmaxf(tmax, __shfl_xor(tmax, 16));
    float mnew = fmaxf(mrow, tmax);
    float corr = __expf(mrow - mnew);
    float p[16];
    float psum = 0.0f;
#pragma unroll
    for (int j = 0; j < 16; ++j) {
      p[j] = __expf(sv[j] - mnew);
      psum += p[j];
    }
    psum += __shfl_xor(psum, 16);
    lrow = lrow * corr + psum;
    mrow = mnew;
#pragma unroll
    for (int v = 0; v < 8; ++v) { o0[v] *= corr; o1[v] *= corr; }

    // P^T as B-fragment: one half-exchange with the partner lane
    float ex[8];
#pragma unroll
    for (int v = 0; v < 8; ++v) ex[v] = __shfl_xor(hh ? p[v] : p[v + 8], 16);
    v16bf bp;
#pragma unroll
    for (int j = 0; j < 8; ++j) bp[j] = (bf16_t)(hh ? ex[j] : p[j]);
#pragma unroll
    for (int j = 0; j < 8; ++j) bp[j + 8] = (bf16_t)(hh ? p[8 + j] : ex[j]);

    // O^T += V^T * P^T  (two A-fragments cover d = 0..15 and 16..31)
    v16bf av0 = load_fragA(&sV[cur][laneLo * 40], hh);
    v16bf av1 = load_fragA(&sV[cur][(16 + laneLo) * 40], hh);
    o0 = wmma_bf16(av0, bp, o0);
    o1 = wmma_bf16(av1, bp, o1);

    cur ^= 1;
  }

  const float inv = 1.0f / lrow;
  bf16_t* orow = Obf + ((size_t)b * HWN + n) * DMODEL + h * DK;
#pragma unroll
  for (int v = 0; v < 8; ++v) {
    int d0 = v + hh * 8;
    orow[d0]      = (bf16_t)(o0[v] * inv);
    orow[16 + d0] = (bf16_t)(o1[v] * inv);
  }
}

// ---------------------------------------------------------------- output GEMM
// y = x + gamma * (Wo · O) + bo
__global__ void __launch_bounds__(128)
out_kernel(const bf16_t* __restrict__ Obf, const bf16_t* __restrict__ Wob,
           const float* __restrict__ bo, const float* __restrict__ x,
           const float* __restrict__ gamma_p, float* __restrict__ y) {
  const int lane = threadIdx.x & 31;
  const int laneLo = lane & 15;
  const int hh = lane >> 4;
  const int tile = blockIdx.x * 4 + (threadIdx.x >> 5); // 0..4607
  const int b = tile / (16 * 144);
  const int rem = tile % (16 * 144);
  const int o0 = (rem / 144) * 16;
  const int n0 = (rem % 144) * 16;
  const float gamma = *gamma_p;

  const int n = n0 + laneLo;
  const int oA = o0 + laneLo;
  const bf16_t* orow = Obf + ((size_t)b * HWN + n) * DMODEL;

  v8f acc = {};
  for (int c0 = 0; c0 < DMODEL; c0 += 32) {
    v16bf bfrag = load_fragB(orow + c0, hh);
    v16bf wfrag = load_fragA(Wob + (size_t)oA * DMODEL + c0, hh);
    acc = wmma_bf16(wfrag, bfrag, acc);
  }
#pragma unroll
  for (int v = 0; v < 8; ++v) {
    int o = o0 + v + hh * 8;
    size_t idx = ((size_t)b * CCH + o) * HWN + n;
    y[idx] = x[idx] + gamma * acc[v] + bo[o];
  }
}

// ---------------------------------------------------------------- launcher
extern "C" void kernel_launch(void* const* d_in, const int* in_sizes, int n_in,
                              void* d_out, int out_size, void* d_ws, size_t ws_size,
                              hipStream_t stream) {
  (void)in_sizes; (void)n_in; (void)out_size; (void)ws_size;
  const float* x   = (const float*)d_in[0];
  const float* Wq  = (const float*)d_in[1];
  const float* bq  = (const float*)d_in[2];
  const float* Wk  = (const float*)d_in[3];
  const float* bk  = (const float*)d_in[4];
  const float* Wv  = (const float*)d_in[5];
  const float* bv  = (const float*)d_in[6];
  const float* Wo  = (const float*)d_in[7];
  const float* bo  = (const float*)d_in[8];
  const float* lam = (const float*)d_in[9];
  const float* gam = (const float*)d_in[10];
  float* y = (float*)d_out;

  // workspace carve-up (256B aligned)
  char* p = (char*)d_ws;
  size_t off = 0;
  auto take = [&](size_t bytes) {
    void* r = p + off;
    off += (bytes + 255) & ~(size_t)255;
    return r;
  };
  float*  pe  = (float*)take((size_t)DMODEL * HWN * 4);
  float*  dm  = (float*)take((size_t)HWN * HWN * 4);
  bf16_t* Wqb = (bf16_t*)take((size_t)DMODEL * CCH * 2);
  bf16_t* Wkb = (bf16_t*)take((size_t)DMODEL * CCH * 2);
  bf16_t* Wvb = (bf16_t*)take((size_t)DMODEL * CCH * 2);
  bf16_t* Wob = (bf16_t*)take((size_t)CCH * DMODEL * 2);
  bf16_t* xbf = (bf16_t*)take((size_t)BB * HWN * CCH * 2);
  bf16_t* Qbf = (bf16_t*)take((size_t)BB * NHEADS * HWN * DK * 2);
  bf16_t* Kbf = (bf16_t*)take((size_t)BB * NHEADS * HWN * DK * 2);
  bf16_t* Vbf = (bf16_t*)take((size_t)BB * DMODEL * HWN * 2);
  bf16_t* Obf = (bf16_t*)take((size_t)BB * HWN * DMODEL * 2); // total ~36 MB

  pe_kernel<<<(DMODEL * HWN + 255) / 256, 256, 0, stream>>>(pe);
  dm_kernel<<<(HWN * HWN + 255) / 256, 256, 0, stream>>>(dm);
  cvt_bf16_kernel<<<256, 256, 0, stream>>>(Wq, Wqb, DMODEL * CCH);
  cvt_bf16_kernel<<<256, 256, 0, stream>>>(Wk, Wkb, DMODEL * CCH);
  cvt_bf16_kernel<<<256, 256, 0, stream>>>(Wv, Wvb, DMODEL * CCH);
  cvt_bf16_kernel<<<256, 256, 0, stream>>>(Wo, Wob, CCH * DMODEL);
  transpose_x_kernel<<<(BB * HWN * CCH + 255) / 256, 256, 0, stream>>>(x, xbf);

  proj_kernel<<<1152, 128, 0, stream>>>(xbf, Wqb, Wkb, Wvb, bq, bk, bv, pe,
                                        Qbf, Kbf, Vbf);
  attn_kernel<<<576, 128, 0, stream>>>(Qbf, Kbf, Vbf, dm, lam, Obf);
  out_kernel<<<1152, 128, 0, stream>>>(Obf, Wob, bo, x, gam, y);
}
